// RelativeAttention_90778428768990
// MI455X (gfx1250) — matrix operand order repaired and checked by reading, not compiled, not run on hardware
//
#include <hip/hip_runtime.h>
#include <stdint.h>

typedef _Float16 half_t;
typedef __attribute__((ext_vector_type(16))) _Float16 v16h;
typedef __attribute__((ext_vector_type(8)))  _Float16 v8h;
typedef __attribute__((ext_vector_type(8)))  float    v8f;
typedef __attribute__((ext_vector_type(2)))  float    v2f;

#define WH 7
#define WW 7
#define L  49
#define C  128
#define H  32
#define BN 4096
#define LP 64          // L padded to 64 rows
#define QKS 392        // qkv LDS row stride (halfs), 16B aligned (392*2=784=49*16)

// shared memory carve (bytes):
//   [0,     16384)  xbuf   f16 [64][128]   (x in phase 0/1, attn-out in 2/3)
//   [16384, 66560)  qkvs   f16 [64][392]   -- overlaid with xstage f32 [49][128]
//   [66560, 82944)  pbuf   f16 [8][16][64]
#define SM_XBUF   0
#define SM_QKVS   16384
#define SM_PBUF   66560
#define SM_TOTAL  82944

// workspace layout (bytes):
#define WS_WQKVT   0        // f16 [384][128] (K-contiguous per output col)
#define WS_WPROJT  98304    // f16 [128][128]
#define WS_BIAS    131072   // f32 [32][49][49]

// ---------------------------------------------------------------- prep ----
__global__ void swin_prep_kernel(const float* __restrict__ Wqkv,
                                 const float* __restrict__ Wproj,
                                 const float* __restrict__ bias_table,
                                 half_t* __restrict__ WqkvT,
                                 half_t* __restrict__ WprojT,
                                 float* __restrict__ biasExp) {
  int i = blockIdx.x * blockDim.x + threadIdx.x;
  if (i < 384 * 128) {
    int n = i >> 7, k = i & 127;
    WqkvT[n * 128 + k] = (half_t)Wqkv[k * 384 + n];
  }
  int j = i - 384 * 128;
  if (j >= 0 && j < 128 * 128) {
    int n = j >> 7, k = j & 127;
    WprojT[n * 128 + k] = (half_t)Wproj[k * 128 + n];
  }
  int m = i - 384 * 128 - 128 * 128;
  if (m >= 0 && m < H * L * L) {
    int h = m / (L * L);
    int rem = m % (L * L);
    int q = rem / L, s = rem % L;
    int thq = q / WW, twq = q % WW;
    int ths = s / WW, tws = s % WW;
    int idx = (thq - ths + WH - 1) * (2 * WW - 1) + (twq - tws + WW - 1);
    biasExp[m] = bias_table[idx * H + h];
  }
}

// ------------------------------------------------------- fragment loaders --
// A-matrix f16 16x32 fragment from row-major f16 array (stride in halfs).
// Lane l: M=l&15, K = klo..klo+15, klo = l<16 ? 0 : 16.
__device__ inline v16h load_a_f16(const half_t* base, int stride) {
  int l = threadIdx.x & 31;
  const half_t* p = base + (l & 15) * stride + ((l < 16) ? 0 : 16);
  v8h lo = *(const v8h*)p;
  v8h hi = *(const v8h*)(p + 8);
  v16h r;
#pragma unroll
  for (int e = 0; e < 8; ++e) { r[e] = lo[e]; r[8 + e] = hi[e]; }
  return r;
}

// B-matrix f16 32x16 fragment from W stored [N][128] (K-contiguous).
// Lane l: N = ncolBase + (l&15); elements 0..15 = K khalf..khalf+15, khalf = l<16?0:16.
__device__ inline v16h load_b_f16(const half_t* __restrict__ W, int ncolBase, int kbase) {
  int l = threadIdx.x & 31;
  const half_t* p = W + (ncolBase + (l & 15)) * 128 + kbase + ((l < 16) ? 0 : 16);
  v8h b0 = *(const v8h*)p;
  v8h b1 = *(const v8h*)(p + 8);
  v16h r;
#pragma unroll
  for (int e = 0; e < 8; ++e) { r[e] = b0[e]; r[8 + e] = b1[e]; }
  return r;
}

// ---------------------------------------------------------------- main ----
__global__ __launch_bounds__(256) void swin_attn_kernel(
    const float* __restrict__ x, const float* __restrict__ b_qkv,
    const float* __restrict__ b_proj, const half_t* __restrict__ WqkvT,
    const half_t* __restrict__ WprojT, const float* __restrict__ biasExp,
    float* __restrict__ out) {
  __shared__ __align__(128) char smem[SM_TOTAL];
  half_t* xbuf   = (half_t*)(smem + SM_XBUF);
  half_t* qkvs   = (half_t*)(smem + SM_QKVS);
  float*  xstage = (float*)(smem + SM_QKVS);   // overlaid: dead before phase 1
  half_t* pbase  = (half_t*)(smem + SM_PBUF);

  const int tid = threadIdx.x;
  const int wave = tid >> 5;
  const int lane = tid & 31;
  const int l15 = lane & 15;
  const int hi8 = (lane >> 4) << 3;        // 0 for lanes 0-15, 8 for 16-31
  const int b = blockIdx.x;

  // ---- Phase 0a: async-copy x[b] (f32) straight into LDS staging -------
  {
    const uint32_t lds_base = (uint32_t)(uintptr_t)xstage;  // LDS offset in addr[31:0]
    const float* gbase = x + (size_t)b * (L * C);
#pragma unroll 2
    for (int i = tid; i < (L * C) / 4; i += 256) {          // 16B per lane
      uint32_t lds_off = lds_base + (uint32_t)i * 16u;
      const float* g = gbase + i * 4;
      asm volatile("global_load_async_to_lds_b128 %0, %1, off"
                   :: "v"(lds_off), "v"(g) : "memory");
    }
    asm volatile("s_wait_asynccnt 0" ::: "memory");
  }
  __syncthreads();

  // ---- Phase 0b: f32 -> f16 convert, zero-pad rows 49..63 --------------
  for (int i = tid; i < LP * C; i += 256) {
    int row = i >> 7;
    float v = (row < L) ? xstage[i] : 0.f;
    xbuf[i] = (half_t)v;
  }
  __syncthreads();

  // ---- Phase 1: qkv[64,384] = x @ Wqkv + b_qkv (f16 WMMA) --------------
  for (int nt = wave * 3; nt < wave * 3 + 3; ++nt) {
    const int ncol = nt * 16 + l15;
    const float bias = b_qkv[ncol];
    v16h bfr[4];
#pragma unroll
    for (int ks = 0; ks < 4; ++ks) bfr[ks] = load_b_f16(WqkvT, nt * 16, ks * 32);
    for (int mt = 0; mt < 4; ++mt) {
      v8f acc = {};
#pragma unroll
      for (int ks = 0; ks < 4; ++ks) {
        v16h a = load_a_f16(xbuf + mt * 16 * C + ks * 32, C);
        acc = __builtin_amdgcn_wmma_f32_16x16x32_f16(false, a, false, bfr[ks],
                                                     (short)0, acc, false, false);
      }
#pragma unroll
      for (int r = 0; r < 8; ++r)
        qkvs[(mt * 16 + r + hi8) * QKS + ncol] = (half_t)(acc[r] + bias);
    }
  }
  __syncthreads();

  // ---- Phase 2: attention, 4 heads per wave ----------------------------
  const float scale = 0.5f;  // HD^-0.5
  for (int h = wave; h < H; h += 8) {
    // v B-fragments (f16, K=64 in two 32-steps), hoisted per head.
    v16h vfrag[2];
    {
      int khalf = (lane < 16) ? 0 : 16;
      int vcol = 256 + h * 4 + (l15 & 3);
#pragma unroll
      for (int ks = 0; ks < 2; ++ks) {
        v16h r;
#pragma unroll
        for (int e = 0; e < 16; ++e)
          r[e] = qkvs[(ks * 32 + khalf + e) * QKS + vcol];
        vfrag[ks] = r;
      }
    }
    for (int mt = 0; mt < 4; ++mt) {
      // q A-frag (f32 16x4): lane lo half K={0,1}, hi half K={2,3}
      v2f aq;
      {
        int m = mt * 16 + l15;
        int d0 = (lane < 16) ? 0 : 2;
        aq.x = scale * (float)qkvs[m * QKS + h * 4 + d0];
        aq.y = scale * (float)qkvs[m * QKS + h * 4 + d0 + 1];
      }
      float p[4][8];
      float rmax[8], rsum[8];
#pragma unroll
      for (int r = 0; r < 8; ++r) rmax[r] = -3.0e38f;
#pragma unroll
      for (int nt = 0; nt < 4; ++nt) {
        v2f bk;
        {
          int s = nt * 16 + l15;
          int d0 = (lane < 16) ? 0 : 2;
          bk.x = (float)qkvs[s * QKS + 128 + h * 4 + d0];
          bk.y = (float)qkvs[s * QKS + 128 + h * 4 + d0 + 1];
        }
        v8f zero = {};
        v8f acc = __builtin_amdgcn_wmma_f32_16x16x4_f32(false, aq, false, bk,
                                                        (short)0, zero, false, false);
#pragma unroll
        for (int r = 0; r < 8; ++r) {
          int row = mt * 16 + r + hi8;
          int col = nt * 16 + l15;
          float v = (row < L && col < L)
                        ? acc[r] + biasExp[h * (L * L) + row * L + col]
                        : -3.0e38f;
          p[nt][r] = v;
          rmax[r] = fmaxf(rmax[r], v);
        }
      }
      // row max/sum across the 16 lanes holding each row
#pragma unroll
      for (int r = 0; r < 8; ++r) {
        float m = rmax[r];
        m = fmaxf(m, __shfl_xor(m, 1));
        m = fmaxf(m, __shfl_xor(m, 2));
        m = fmaxf(m, __shfl_xor(m, 4));
        m = fmaxf(m, __shfl_xor(m, 8));
        rmax[r] = m;
        rsum[r] = 0.f;
      }
#pragma unroll
      for (int nt = 0; nt < 4; ++nt)
#pragma unroll
        for (int r = 0; r < 8; ++r) {
          float e = __expf(p[nt][r] - rmax[r]);
          p[nt][r] = e;
          rsum[r] += e;
        }
#pragma unroll
      for (int r = 0; r < 8; ++r) {
        float s = rsum[r];
        s += __shfl_xor(s, 1);
        s += __shfl_xor(s, 2);
        s += __shfl_xor(s, 4);
        s += __shfl_xor(s, 8);
        rsum[r] = s;
      }
      // write P (f16) to per-wave scratch, read back as A-frags
      half_t* pb = pbase + wave * (16 * 64);
#pragma unroll
      for (int nt = 0; nt < 4; ++nt)
#pragma unroll
        for (int r = 0; r < 8; ++r)
          pb[(r + hi8) * 64 + nt * 16 + l15] = (half_t)p[nt][r];
      v8f accO = {};
#pragma unroll
      for (int ks = 0; ks < 2; ++ks) {
        v16h pa = load_a_f16(pb + ks * 32, 64);
        accO = __builtin_amdgcn_wmma_f32_16x16x32_f16(false, pa, false, vfrag[ks],
                                                      (short)0, accO, false, false);
      }
      if (l15 < 4) {
#pragma unroll
        for (int r = 0; r < 8; ++r)
          xbuf[(mt * 16 + r + hi8) * C + h * 4 + l15] =
              (half_t)(accO[r] / rsum[r]);
      }
    }
  }
  __syncthreads();

  // ---- Phase 3: out = attnOut @ Wproj + b_proj -------------------------
  {
    const int ncol = wave * 16 + l15;
    const float bp = b_proj[ncol];
    v16h bfr[4];
#pragma unroll
    for (int ks = 0; ks < 4; ++ks) bfr[ks] = load_b_f16(WprojT, wave * 16, ks * 32);
    for (int mt = 0; mt < 4; ++mt) {
      v8f acc = {};
#pragma unroll
      for (int ks = 0; ks < 4; ++ks) {
        v16h a = load_a_f16(xbuf + mt * 16 * C + ks * 32, C);
        acc = __builtin_amdgcn_wmma_f32_16x16x32_f16(false, a, false, bfr[ks],
                                                     (short)0, acc, false, false);
      }
#pragma unroll
      for (int r = 0; r < 8; ++r) {
        int row = mt * 16 + r + hi8;
        if (row < L) out[(size_t)b * (L * C) + row * C + ncol] = acc[r] + bp;
      }
    }
  }
}

// -------------------------------------------------------------- launch ----
extern "C" void kernel_launch(void* const* d_in, const int* in_sizes, int n_in,
                              void* d_out, int out_size, void* d_ws, size_t ws_size,
                              hipStream_t stream) {
  (void)in_sizes; (void)n_in; (void)out_size; (void)ws_size;
  const float* x          = (const float*)d_in[0];
  const float* Wqkv       = (const float*)d_in[1];
  const float* bqkv       = (const float*)d_in[2];
  const float* Wproj      = (const float*)d_in[3];
  const float* bproj      = (const float*)d_in[4];
  const float* bias_table = (const float*)d_in[5];

  char* ws = (char*)d_ws;
  half_t* WqkvT  = (half_t*)(ws + WS_WQKVT);
  half_t* WprojT = (half_t*)(ws + WS_WPROJT);
  float*  biasEx = (float*)(ws + WS_BIAS);

  int prep_n = 384 * 128 + 128 * 128 + H * L * L;
  swin_prep_kernel<<<(prep_n + 255) / 256, 256, 0, stream>>>(
      Wqkv, Wproj, bias_table, WqkvT, WprojT, biasEx);
  swin_attn_kernel<<<BN, 256, 0, stream>>>(x, bqkv, bproj, WqkvT, WprojT,
                                           biasEx, (float*)d_out);
}
